// TRIP_86268713107801
// MI455X (gfx1250) — matrix operand order repaired and checked by reading, not compiled, not run on hardware
//
#include <hip/hip_runtime.h>

typedef _Float16 v16h __attribute__((ext_vector_type(16)));
typedef _Float16 v8h  __attribute__((ext_vector_type(8)));
typedef float    v8f  __attribute__((ext_vector_type(8)));

__device__ __forceinline__ float softplus_f(float x) {
    return (x > 20.0f) ? x : log1pf(__expf(x));
}

// ---------------------------------------------------------------------------
// Kernel 1: sp_cores = softplus(cores) -> f16, stored in WMMA B-fragment order:
// flat = ((((i*64 + nt)*8 + kt)*32 + lane)*16 + h)
//   lane 0-15:  column n = lane,     K = kt*32 + (h<8 ? h : 16 + h-8)
//   lane 16-31: column n = lane-16,  K = kt*32 + (h<8 ? 8 + h : 24 + h-8)
// ---------------------------------------------------------------------------
__global__ void trip_prep_cores(const float* __restrict__ cores,
                                _Float16* __restrict__ coresF16) {
    int flat = blockIdx.x * 256 + threadIdx.x;     // 2,097,152 total
    int h    = flat & 15;
    int lane = (flat >> 4) & 31;
    int kt   = (flat >> 9) & 7;
    int nt   = (flat >> 12) & 63;
    int i    = flat >> 18;
    int n    = lane & 15;
    int kh   = lane >> 4;
    int d    = (kt << 5) + ((h >> 3) << 4) + (kh << 3) + (h & 7); // K index == d (0..255)
    int j    = (nt << 4) + n;                      // flattened (r1,r2) column
    int r1   = j >> 5, r2 = j & 31;
    float x  = cores[(((i << 8) + d) << 10) + (r1 << 5) + r2];
    coresF16[flat] = (_Float16)softplus_f(x);
}

// ---------------------------------------------------------------------------
// Kernel 2: per-(d,i) Gaussian constants so the hot loop has one v_exp each:
// invs = exp(-ls), coef = exp(-0.5*log2pi - ls)
// ---------------------------------------------------------------------------
__global__ void trip_prep_gauss(const float* __restrict__ log_scale,
                                float* __restrict__ invs,
                                float* __restrict__ coef) {
    int t = blockIdx.x * 256 + threadIdx.x;        // 256*8 = 2048
    float ls = log_scale[t];
    invs[t] = __expf(-ls);
    coef[t] = __expf(-0.91893853320467274f - ls);  // 0.5*log(2*pi)
}

// ---------------------------------------------------------------------------
// Kernel 3: normalizer chain  lognorm = log(trace(prod_i sum_d sp_core_i[d])).
// ---------------------------------------------------------------------------
__global__ void trip_prep_norm(const float* __restrict__ cores,
                               float* __restrict__ lognorm) {
    __shared__ float cLDS[1024];
    __shared__ float sLDS[1024];
    int tid = threadIdx.x;
    int r1 = tid >> 5, r2 = tid & 31;

    float acc = 0.0f;
    for (int d = 0; d < 256; ++d)
        acc += softplus_f(cores[(d << 10) + tid]);
    cLDS[tid] = acc;
    __syncthreads();

    for (int i = 1; i < 8; ++i) {
        float s = 0.0f;
        for (int d = 0; d < 256; ++d)
            s += softplus_f(cores[(((i << 8) + d) << 10) + tid]);
        sLDS[tid] = s;
        __syncthreads();
        float nv = 0.0f;
        for (int k = 0; k < 32; ++k)
            nv += cLDS[(r1 << 5) + k] * sLDS[(k << 5) + r2];
        __syncthreads();
        cLDS[tid] = nv;
        __syncthreads();
    }
    if (tid == 0) {
        float tr = 0.0f;
        for (int r = 0; r < 32; ++r) tr += cLDS[r * 33];
        *lognorm = logf(tr);
    }
}

// ---------------------------------------------------------------------------
// Margin GEMM: 8 n-tiles per wave, K = 256 (8 WMMA k-steps).
// TO_PROB selects the compile-time store layout:
//   true : prob row-major      addr = m*1024 + j
//   false: marginT [m][c][k]   addr = m*1024 + (j&31)*32 + (j>>5)
// ---------------------------------------------------------------------------
template <bool TO_PROB>
__device__ __forceinline__ void margin_gemm(const v16h* __restrict__ cbase,
                                            const v16h afrag[8],
                                            _Float16* __restrict__ dst,
                                            int wave, int lane, int g, int nlo) {
    #pragma unroll
    for (int q = 0; q < 8; ++q) {
        int nt = (wave << 3) + q;
        v8f acc = {};
        #pragma unroll
        for (int kt = 0; kt < 8; ++kt) {
            v16h bfrag = cbase[(((nt << 3) + kt) << 5) + lane];
            acc = __builtin_amdgcn_wmma_f32_16x16x32_f16(
                false, afrag[kt], false, bfrag, (short)0, acc, false, false);
        }
        // C layout: VGPR v -> M = g*8+v (batch row m), N = nlo; j = nt*16+nlo
        int j   = (nt << 4) + nlo;
        int off = TO_PROB ? j : (((j & 31) << 5) | (j >> 5));
        #pragma unroll
        for (int v = 0; v < 8; ++v)
            dst[(((g << 3) + v) << 10) + off] = (_Float16)acc[v];
    }
}

// ---------------------------------------------------------------------------
// Main fused kernel: 16 batch rows per block, 8 waves.
// LDS (dynamic, ~72.5KB):
//   pLDS[16][256] f16      Gaussian weights (A operand source)
//   marginT[16][1024] f16  M_b^T row-major [b][c][k]
//   prob[16][1024]    f16  P_b   row-major [b][r][k]
//   vLDS[16][8]       f32  block's value tile
// Chain update computed as prob_new^T = margin^T @ prob^T so every LDS access
// in the update is an aligned vector op.
// ---------------------------------------------------------------------------
__global__ void __launch_bounds__(256)
trip_main(const float* __restrict__ value,
          const float* __restrict__ location,
          const float* __restrict__ invs,
          const float* __restrict__ coef,
          const _Float16* __restrict__ coresF16,
          const float* __restrict__ lognorm,
          float* __restrict__ out) {
    extern __shared__ _Float16 smem[];
    _Float16* pLDS    = smem;                           // 16*256  = 8KB
    _Float16* marginT = smem + 16 * 256;                // 16*1024 = 32KB
    _Float16* prob    = smem + 16 * 256 + 16 * 1024;    // 16*1024 = 32KB
    float*    vLDS    = (float*)(smem + 16 * 256 + 2 * 16 * 1024); // 512B

    const int tid  = threadIdx.x;
    const int lane = tid & 31;
    const int wave = tid >> 5;
    const int b0   = blockIdx.x << 4;
    const int g    = lane >> 4;       // lane half-group
    const int nlo  = lane & 15;

    // block's 16x8 value tile -> LDS (read 16x per mode afterwards)
    if (tid < 128)
        vLDS[tid] = value[((b0 + (tid >> 3)) << 3) + (tid & 7)];

    // per-thread Gaussian constants are indexed by d == tid (loop-invariant)
    const float loc0 = location[(tid << 3) + 0]; // mode 0 handled peeled below

    // ---------------- mode 0 (peeled: margin_0 -> prob directly) ----------
    __syncthreads();
    {
        const float is = invs[(tid << 3)], cf = coef[(tid << 3)];
        #pragma unroll
        for (int t = 0; t < 16; ++t) {
            float z = (vLDS[t << 3] - loc0) * is;
            pLDS[(t << 8) + tid] = (_Float16)(cf * __expf(-0.5f * z * z));
        }
        __syncthreads();

        v16h afrag[8];
        #pragma unroll
        for (int kt = 0; kt < 8; ++kt) {
            int base = (nlo << 8) + (kt << 5) + (g << 3);
            v8h lo = *(const v8h*)&pLDS[base];
            v8h hi = *(const v8h*)&pLDS[base + 16];
            afrag[kt] = __builtin_shufflevector(lo, hi,
                0, 1, 2, 3, 4, 5, 6, 7, 8, 9, 10, 11, 12, 13, 14, 15);
        }
        margin_gemm<true>((const v16h*)coresF16, afrag, prob, wave, lane, g, nlo);
        __syncthreads();
    }

    // ---------------- modes 1..7 ------------------------------------------
    for (int i = 1; i < 8; ++i) {
        // (1) Gaussian weights p[m][d], one transcendental each
        {
            const float lc = location[(tid << 3) + i];
            const float is = invs[(tid << 3) + i];
            const float cf = coef[(tid << 3) + i];
            #pragma unroll
            for (int t = 0; t < 16; ++t) {
                float z = (vLDS[(t << 3) + i] - lc) * is;
                pLDS[(t << 8) + tid] = (_Float16)(cf * __expf(-0.5f * z * z));
            }
        }
        __syncthreads();

        // (2) A fragments (16x32 per K-step) from LDS p
        v16h afrag[8];
        #pragma unroll
        for (int kt = 0; kt < 8; ++kt) {
            int base = (nlo << 8) + (kt << 5) + (g << 3);
            v8h lo = *(const v8h*)&pLDS[base];
            v8h hi = *(const v8h*)&pLDS[base + 16];
            afrag[kt] = __builtin_shufflevector(lo, hi,
                0, 1, 2, 3, 4, 5, 6, 7, 8, 9, 10, 11, 12, 13, 14, 15);
        }

        // (3) marginT = (p @ sp_core_i)^T-layout : 8 n-tiles per wave
        const v16h* cbase = (const v16h*)(coresF16 + ((size_t)i << 18));
        if (i < 7)
            __builtin_prefetch(coresF16 + (((size_t)(i + 1)) << 18) +
                               (((size_t)wave << 3) << 12), 0, 1);
        margin_gemm<false>(cbase, afrag, marginT, wave, lane, g, nlo);
        __syncthreads();

        // (4) prob_new^T = margin^T @ prob^T, per batch element (32x32x32)
        #pragma unroll
        for (int bi = 0; bi < 2; ++bi) {
            int bb = (wave << 1) + bi;
            _Float16* pb = prob    + (bb << 10);
            _Float16* mb = marginT + (bb << 10);
            // A fragments: rows c of margin^T (contiguous in k)
            v16h ma[2];
            #pragma unroll
            for (int mt = 0; mt < 2; ++mt) {
                int base = (((mt << 4) + nlo) << 5) + (g << 4);
                v8h lo = *(const v8h*)&mb[base];
                v8h hi = *(const v8h*)&mb[base + 8];
                ma[mt] = __builtin_shufflevector(lo, hi,
                    0, 1, 2, 3, 4, 5, 6, 7, 8, 9, 10, 11, 12, 13, 14, 15);
            }
            // B fragments: prob^T[k][k'] = prob[k'][k] -> one aligned 32B load
            v16h pbf[2];
            #pragma unroll
            for (int ntt = 0; ntt < 2; ++ntt)
                pbf[ntt] = *(const v16h*)&pb[(((ntt << 4) + nlo) << 5) + (g << 4)];

            v8f accs[4];
            #pragma unroll
            for (int ntt = 0; ntt < 2; ++ntt)
                #pragma unroll
                for (int mt = 0; mt < 2; ++mt) {
                    v8f cz = {};
                    accs[mt * 2 + ntt] = __builtin_amdgcn_wmma_f32_16x16x32_f16(
                        false, ma[mt], false, pbf[ntt], (short)0, cz, false, false);
                }
            // D[c][k'] = prob_new[k'][c]: contiguous in v -> packed 128b store
            #pragma unroll
            for (int mt = 0; mt < 2; ++mt)
                #pragma unroll
                for (int ntt = 0; ntt < 2; ++ntt) {
                    v8h hv;
                    #pragma unroll
                    for (int v = 0; v < 8; ++v)
                        hv[v] = (_Float16)accs[mt * 2 + ntt][v];
                    *(v8h*)&pb[(((ntt << 4) + nlo) << 5) + (mt << 4) + (g << 3)] = hv;
                }
        }
        __syncthreads();
    }

    // ---- (5) trace + log
    if (tid < 16) {
        const _Float16* pb = prob + (tid << 10);
        float tr = 0.0f;
        #pragma unroll
        for (int r = 0; r < 32; ++r) tr += (float)pb[r * 33];
        out[b0 + tid] = logf(fmaxf(tr, 1e-12f)) - *lognorm;
    }
}

// ---------------------------------------------------------------------------
extern "C" void kernel_launch(void* const* d_in, const int* in_sizes, int n_in,
                              void* d_out, int out_size, void* d_ws, size_t ws_size,
                              hipStream_t stream) {
    const float* value     = (const float*)d_in[0];   // (32768, 8)
    const float* location  = (const float*)d_in[1];   // (256, 8)
    const float* log_scale = (const float*)d_in[2];   // (256, 8)
    const float* cores     = (const float*)d_in[3];   // (8, 256, 32, 32)
    float* out = (float*)d_out;                       // (32768,)

    char* ws = (char*)d_ws;
    _Float16* coresF16 = (_Float16*)ws;                            // 4 MB
    float* lognorm = (float*)(ws + (size_t)4194304);               // 64 B slot
    float* invs    = (float*)(ws + (size_t)4194304 + 64);          // 8 KB
    float* coef    = (float*)(ws + (size_t)4194304 + 64 + 8192);   // 8 KB

    trip_prep_cores<<<8192, 256, 0, stream>>>(cores, coresF16);
    trip_prep_gauss<<<8, 256, 0, stream>>>(log_scale, invs, coef);
    trip_prep_norm<<<1, 1024, 0, stream>>>(cores, lognorm);

    const size_t ldsBytes =
        (16 * 256 + 2 * 16 * 1024) * sizeof(_Float16) + 16 * 8 * sizeof(float); // 74240
    trip_main<<<32768 / 16, 256, ldsBytes, stream>>>(
        value, location, invs, coef, coresF16, lognorm, out);
}
